// VectorQuantizer_24550033064109
// MI455X (gfx1250) — matrix operand order repaired and checked by reading, not compiled, not run on hardware
//
#include <hip/hip_runtime.h>

#define NUM_CODES 512
#define DIM       64
#define NPIX      (32*64*64)                 // 131072 pixels
#define WAVES     8
#define STRIPS    4
#define ROWS_PER_BLOCK (WAVES*STRIPS*16)     // 512 pixels per block
#define EPAD      72                          // halves per LDS codebook row (bank pad)
#define ELDS_BYTES (NUM_CODES*EPAD*2)         // 73728
#define SMEM_BYTES (ELDS_BYTES + NUM_CODES*4) // + fp32 norms = 75776

typedef __attribute__((ext_vector_type(16))) _Float16 v16h;
typedef __attribute__((ext_vector_type(8)))  _Float16 v8h;
typedef __attribute__((ext_vector_type(4)))  _Float16 v4h;
typedef __attribute__((ext_vector_type(8)))  float    v8f;
typedef __attribute__((ext_vector_type(4)))  float    v4f;
typedef __attribute__((ext_vector_type(2)))  float    v2f;

__global__ __launch_bounds__(256) void vq_wmma_kernel(
    const float* __restrict__ X, const float* __restrict__ E,
    int* __restrict__ codes, float* __restrict__ vecs)
{
    extern __shared__ char smem[];
    _Float16* Elds  = (_Float16*)smem;                 // [512][72] f16, padded rows
    float*    eNorm = (float*)(smem + ELDS_BYTES);     // [512] fp32 ||e||^2

    const int t = threadIdx.x;

    // ---- Stage codebook into LDS as f16 (coalesced float4 reads, L2-resident) ----
    for (int f = t; f < NUM_CODES*(DIM/4); f += 256) {
        int row = f >> 4;
        int c4  = (f & 15) << 2;
        v4f v = *(const v4f*)(E + row*DIM + c4);
        v4h h;
        h[0]=(_Float16)v.x; h[1]=(_Float16)v.y; h[2]=(_Float16)v.z; h[3]=(_Float16)v.w;
        *(v4h*)(Elds + row*EPAD + c4) = h;
    }
    // ---- Exact fp32 code norms (deterministic order) ----
    for (int r = t; r < NUM_CODES; r += 256) {
        const float* er = E + r*DIM;
        float s = 0.f;
        #pragma unroll
        for (int k = 0; k < DIM; k += 4) {
            v4f v = *(const v4f*)(er + k);
            s += v.x*v.x + v.y*v.y + v.z*v.z + v.w*v.w;
        }
        eNorm[r] = s;
    }
    __syncthreads();

    const int lane = t & 31;
    const int wave = t >> 5;
    const int m    = lane & 15;   // A row within strip / B column (N) within chunk
    const int hi   = lane >> 4;   // half-wave group
    const int base = blockIdx.x * ROWS_PER_BLOCK + wave * (STRIPS*16);

    // ---- A fragments: 16x32 f16, held in registers for all 4 strips ----
    // ISA layout: lanes 0-15 (kbase=0): VGPR0-3 = K 0..7, VGPR4-7 = K 16..23
    //             lanes16-31 (kbase=8): VGPR0-3 = K 8..15, VGPR4-7 = K 24..31
    v16h a[STRIPS][2];
    #pragma unroll
    for (int s = 0; s < STRIPS; ++s) {
        const float* xr = X + (size_t)(base + s*16 + m) * DIM + hi*8;
        #pragma unroll
        for (int h2 = 0; h2 < 2; ++h2) {          // koff = 32*h2
            v16h f;
            #pragma unroll
            for (int g = 0; g < 2; ++g) {          // halves 0..7 -> K+0..7, 8..15 -> K+16..23
                v4f p0 = *(const v4f*)(xr + 32*h2 + 16*g);
                v4f p1 = *(const v4f*)(xr + 32*h2 + 16*g + 4);
                f[8*g+0]=(_Float16)p0.x; f[8*g+1]=(_Float16)p0.y;
                f[8*g+2]=(_Float16)p0.z; f[8*g+3]=(_Float16)p0.w;
                f[8*g+4]=(_Float16)p1.x; f[8*g+5]=(_Float16)p1.y;
                f[8*g+6]=(_Float16)p1.z; f[8*g+7]=(_Float16)p1.w;
            }
            a[s][h2] = f;
        }
    }

    float bestd[STRIPS][8];
    int   besti[STRIPS][8];
    #pragma unroll
    for (int s = 0; s < STRIPS; ++s)
        #pragma unroll
        for (int j = 0; j < 8; ++j) { bestd[s][j] = 3.4e38f; besti[s][j] = 0; }

    // ---- Sweep the codebook: 32 chunks of 16 codes ----
    const int kb = hi << 3;   // B: lanes 0-15 hold K 0..7,16..23, lanes 16-31 K 8..15,24..31 (per 32-chunk)
    for (int c = 0; c < NUM_CODES/16; ++c) {
        const int cbase = c << 4;
        const _Float16* brow = Elds + (cbase + m) * EPAD + kb;
        v16h b[2];
        #pragma unroll
        for (int h2 = 0; h2 < 2; ++h2) {          // 2 x ds_load_b128, K+0..7 and K+16..23
            v8h blo = *(const v8h*)(brow + 32*h2);
            v8h bhi = *(const v8h*)(brow + 32*h2 + 16);
            v16h f;
            #pragma unroll
            for (int i = 0; i < 8; ++i) { f[i] = blo[i]; f[8+i] = bhi[i]; }
            b[h2] = f;
        }
        const float nrm = eNorm[cbase + m];
        #pragma unroll
        for (int s = 0; s < STRIPS; ++s) {
            v8f acc = {};
            acc = __builtin_amdgcn_wmma_f32_16x16x32_f16(false, a[s][0], false, b[0],
                                                         (short)0, acc, false, false);
            acc = __builtin_amdgcn_wmma_f32_16x16x32_f16(false, a[s][1], false, b[1],
                                                         (short)0, acc, false, false);
            #pragma unroll
            for (int j = 0; j < 8; ++j) {
                float d = __builtin_fmaf(-2.0f, acc[j], nrm);
                bool better = d < bestd[s][j];
                bestd[s][j] = better ? d : bestd[s][j];
                besti[s][j] = better ? (cbase + m) : besti[s][j];
            }
        }
    }

    // ---- Argmin across the 16 lanes of each half-wave (C layout: lane = N) ----
    #pragma unroll
    for (int s = 0; s < STRIPS; ++s) {
        #pragma unroll
        for (int j = 0; j < 8; ++j) {
            float d = bestd[s][j];
            int   i = besti[s][j];
            #pragma unroll
            for (int msk = 1; msk < 16; msk <<= 1) {
                float d2 = __shfl_xor(d, msk, 32);
                int   i2 = __shfl_xor(i, msk, 32);
                if (d2 < d || (d2 == d && i2 < i)) { d = d2; i = i2; }
            }
            besti[s][j] = i;                       // all 16 lanes of group now agree
            if (m == 0) codes[base + s*16 + (hi<<3) + j] = i;   // pixel M = j + 8*hi
        }
        // ---- Gather exact fp32 code vectors: whole wave writes 256B per pixel ----
        #pragma unroll
        for (int p = 0; p < 16; ++p) {
            int idx = __shfl(besti[s][p & 7], (p >> 3) << 4, 32);
            v2f ev = *(const v2f*)(E + idx*DIM + 2*lane);
            *(v2f*)(vecs + (size_t)(base + s*16 + p)*DIM + 2*lane) = ev;
        }
    }
}

extern "C" void kernel_launch(void* const* d_in, const int* in_sizes, int n_in,
                              void* d_out, int out_size, void* d_ws, size_t ws_size,
                              hipStream_t stream) {
    const float* X = (const float*)d_in[0];   // [32*64*64][64] fp32
    const float* E = (const float*)d_in[1];   // [512][64] fp32
    int*   codes = (int*)d_out;               // [131072] int32
    float* vecs  = (float*)d_out + NPIX;      // [131072][64] fp32
    dim3 grid(NPIX / ROWS_PER_BLOCK);         // 256 blocks
    vq_wmma_kernel<<<grid, 256, SMEM_BYTES, stream>>>(X, E, codes, vecs);
}
